// BoeNet_14422500180043
// MI455X (gfx1250) — matrix operand order repaired and checked by reading, not compiled, not run on hardware
//
#include <hip/hip_runtime.h>

// ---------------------------------------------------------------------------
// BoeNet on MI455X (gfx1250): the tree expansion is linear, so it collapses to
//   S0 = emb[tok] @ proj_w^T + proj_b
//   S_{l+1} = S_l @ (W_L+W_R)^T + 2^l * (b_L+b_R + s*(sib0+sib1))
//   pooled = (S0 + a1 S1 + a2 S2 + a3 S3) / (1 + 2a1 + 4a2 + 8a3)
//   out = pooled @ out_w^T + out_b            [4096 x 32000]  <- 134 GFLOP
// All GEMMs run through v_wmma_f32_16x16x32_f16 (f16 in, f32 accumulate),
// with double-buffered LDS staging and 64x64 wave tiles (16 wmma : 16 ds_load).
// ---------------------------------------------------------------------------

typedef _Float16 v16h __attribute__((ext_vector_type(16)));
typedef _Float16 v8h  __attribute__((ext_vector_type(8)));
typedef float    v8f  __attribute__((ext_vector_type(8)));

#define NTOK 4096
#define HDIM 512
#define VDIM 32000
#define SIBS 0.044194173824159216f  // 1/sqrt(512)

#define BM 128
#define BN 128
#define BK 32
#define LDST (BK + 8)  // LDS row stride in halves (40 -> 80B, 16B aligned slots)

// ---------------------------------------------------------------------------
// f16 WMMA GEMM: C[M,N] = A[M,K] * W[N,K]^T + bias_scale*bias[N]
// A, W row-major f16. 128 threads = 4 waves, each owns a 64x64 output tile
// (4x4 WMMA accumulators). Double-buffered LDS, one barrier per K-step.
// M%128==N%128==K%32==0.
// ---------------------------------------------------------------------------
template <bool WRITE_F32, bool WRITE_F16>
__global__ __launch_bounds__(128) void gemm_f16_wmma(
    const _Float16* __restrict__ A, const _Float16* __restrict__ W,
    const float* __restrict__ bias, float bias_scale,
    float* __restrict__ Cf, _Float16* __restrict__ Ch, int M, int N, int K) {
  __shared__ _Float16 As[2][BM * LDST];
  __shared__ _Float16 Bs[2][BN * LDST];

  const int tid   = threadIdx.x;   // 0..127
  const int lane  = tid & 31;
  const int wid   = tid >> 5;      // 0..3
  const int waveM = wid >> 1;      // 0..1 : 64-row slab
  const int waveN = wid & 1;       // 0..1 : 64-col slab
  const int blockRow = blockIdx.y * BM;
  const int blockCol = blockIdx.x * BN;

  v8f acc[4][4] = {};

  const int  lhalf  = lane & 15;
  const bool hiHalf = lane >= 16;
  const int  aKoff  = hiHalf ? 8 : 0;   // ISA 16-bit A layout: paired K groups
  const int  bKoff  = hiHalf ? 8 : 0;   // ISA 16-bit B layout: paired K groups (same as A)

  // cooperative staging: thread t streams row t of both tiles (32 halves each)
  const _Float16* aSrc = A + (size_t)(blockRow + tid) * K;
  const _Float16* bSrc = W + (size_t)(blockCol + tid) * K;

  v8h ar[4], br[4];
  const int nk = K / BK;

  // fetch k-step 0
#pragma unroll
  for (int d = 0; d < 4; ++d) {
    ar[d] = *(const v8h*)(aSrc + d * 8);
    br[d] = *(const v8h*)(bSrc + d * 8);
  }
#pragma unroll
  for (int d = 0; d < 4; ++d) {
    *(v8h*)&As[0][tid * LDST + d * 8] = ar[d];
    *(v8h*)&Bs[0][tid * LDST + d * 8] = br[d];
  }
  __syncthreads();

  for (int kk = 0; kk < nk; ++kk) {
    const int cur = kk & 1;
    if (kk + 1 < nk) {
      const int k1 = (kk + 1) * BK;
#pragma unroll
      for (int d = 0; d < 4; ++d) {
        ar[d] = *(const v8h*)(aSrc + k1 + d * 8);
        br[d] = *(const v8h*)(bSrc + k1 + d * 8);
      }
      if (kk + 2 < nk) {  // pull next-next tile line toward L2/WGP$
        __builtin_prefetch(aSrc + (kk + 2) * BK, 0, 1);
        __builtin_prefetch(bSrc + (kk + 2) * BK, 0, 1);
      }
    }

    v16h afrag[4];
#pragma unroll
    for (int i = 0; i < 4; ++i) {
      const _Float16* pa = &As[cur][(waveM * 64 + i * 16 + lhalf) * LDST];
      v8h lo = *(const v8h*)(pa + aKoff);       // K = aKoff .. aKoff+7
      v8h hi = *(const v8h*)(pa + aKoff + 16);  // K = aKoff+16 .. aKoff+23
      afrag[i] = __builtin_shufflevector(lo, hi, 0, 1, 2, 3, 4, 5, 6, 7, 8, 9,
                                         10, 11, 12, 13, 14, 15);
    }
    v16h bfrag[4];
#pragma unroll
    for (int j = 0; j < 4; ++j) {
      const _Float16* pb = &Bs[cur][(waveN * 64 + j * 16 + lhalf) * LDST] + bKoff;
      v8h lo = *(const v8h*)(pb);
      v8h hi = *(const v8h*)(pb + 16);
      bfrag[j] = __builtin_shufflevector(lo, hi, 0, 1, 2, 3, 4, 5, 6, 7, 8, 9,
                                         10, 11, 12, 13, 14, 15);
    }
#pragma unroll
    for (int i = 0; i < 4; ++i)
#pragma unroll
      for (int j = 0; j < 4; ++j)
        acc[i][j] = __builtin_amdgcn_wmma_f32_16x16x32_f16(
            false, afrag[i], false, bfrag[j], (short)0, acc[i][j], false, false);

    if (kk + 1 < nk) {
      const int nxt = 1 - cur;  // last read of buf nxt was iter kk-1 (pre-barrier)
#pragma unroll
      for (int d = 0; d < 4; ++d) {
        *(v8h*)&As[nxt][tid * LDST + d * 8] = ar[d];
        *(v8h*)&Bs[nxt][tid * LDST + d * 8] = br[d];
      }
      __syncthreads();
    }
  }

  // C/D layout: lanes 0-15 hold N=lane, rows r=0..7; lanes 16-31 rows 8..15
  const int rowOff = hiHalf ? 8 : 0;
#pragma unroll
  for (int i = 0; i < 4; ++i) {
    const int rbase = blockRow + waveM * 64 + i * 16 + rowOff;
#pragma unroll
    for (int j = 0; j < 4; ++j) {
      const int col = blockCol + waveN * 64 + j * 16 + lhalf;
      const float b = bias[col] * bias_scale;
#pragma unroll
      for (int r = 0; r < 8; ++r) {
        const float v = acc[i][j][r] + b;
        const size_t idx = (size_t)(rbase + r) * N + col;
        if (WRITE_F32) Cf[idx] = v;
        if (WRITE_F16) Ch[idx] = (_Float16)v;
      }
    }
  }
}

// ---------------------------------------------------------------------------
// Elementwise / prep kernels
// ---------------------------------------------------------------------------
__global__ void f32_to_f16(const float* __restrict__ s, _Float16* __restrict__ d,
                           size_t n) {
  size_t i = (size_t)blockIdx.x * blockDim.x + threadIdx.x;
  size_t stride = (size_t)gridDim.x * blockDim.x;
  for (; i < n; i += stride) d[i] = (_Float16)s[i];
}

__global__ void make_wc(const float* __restrict__ cfw, _Float16* __restrict__ wc) {
  size_t i = (size_t)blockIdx.x * blockDim.x + threadIdx.x;  // 512*512
  wc[i] = (_Float16)(cfw[i] + cfw[i + (size_t)HDIM * HDIM]);
}

__global__ void make_cvec(const float* __restrict__ cfb,
                          const float* __restrict__ sib, float* __restrict__ c) {
  int j = blockIdx.x * blockDim.x + threadIdx.x;  // 512
  c[j] = cfb[j] + cfb[HDIM + j] + SIBS * (sib[j] + sib[HDIM + j]);
}

__global__ void gather_emb(const int* __restrict__ ids,
                           const float* __restrict__ emb,
                           _Float16* __restrict__ A0) {
  size_t i = (size_t)blockIdx.x * blockDim.x + threadIdx.x;  // 4096*512
  int n = (int)(i >> 9), h = (int)(i & 511);
  A0[i] = (_Float16)emb[(size_t)ids[n] * HDIM + h];
}

// LN(S*invM + 0.01*dep) -> X[:, :512] f16; X[:, 512:640] = pde row. Block/row.
__global__ __launch_bounds__(256) void ln_policy_input(
    const float* __restrict__ S, float invM, const float* __restrict__ dep_row,
    const float* __restrict__ g, const float* __restrict__ b,
    const float* __restrict__ pde_row, _Float16* __restrict__ X) {
  __shared__ float red[256];
  const int n = blockIdx.x, tid = threadIdx.x;
  const float v0 = S[(size_t)n * HDIM + tid] * invM + 0.01f * dep_row[tid];
  const float v1 = S[(size_t)n * HDIM + tid + 256] * invM + 0.01f * dep_row[tid + 256];
  red[tid] = v0 + v1;
  __syncthreads();
  for (int s = 128; s > 0; s >>= 1) {
    if (tid < s) red[tid] += red[tid + s];
    __syncthreads();
  }
  const float mean = red[0] / (float)HDIM;
  __syncthreads();
  red[tid] = v0 * v0 + v1 * v1;
  __syncthreads();
  for (int s = 128; s > 0; s >>= 1) {
    if (tid < s) red[tid] += red[tid + s];
    __syncthreads();
  }
  const float var = red[0] / (float)HDIM - mean * mean;
  const float r = rsqrtf(var + 1e-5f);
  X[(size_t)n * 640 + tid]       = (_Float16)((v0 - mean) * r * g[tid] + b[tid]);
  X[(size_t)n * 640 + tid + 256] = (_Float16)((v1 - mean) * r * g[tid + 256] + b[tid + 256]);
  if (tid < 128) X[(size_t)n * 640 + 512 + tid] = (_Float16)pde_row[tid];
}

// relu(LN(Y)) per row; blockDim.x == W (256 or 128)
template <bool WRITE_F32, bool WRITE_F16>
__global__ void ln_relu(const float* __restrict__ Y, const float* __restrict__ g,
                        const float* __restrict__ b, float* __restrict__ Of,
                        _Float16* __restrict__ Oh, int W) {
  __shared__ float red[256];
  const int n = blockIdx.x, tid = threadIdx.x;
  const float v = Y[(size_t)n * W + tid];
  red[tid] = v;
  __syncthreads();
  for (int s = W >> 1; s > 0; s >>= 1) {
    if (tid < s) red[tid] += red[tid + s];
    __syncthreads();
  }
  const float mean = red[0] / (float)W;
  __syncthreads();
  red[tid] = v * v;
  __syncthreads();
  for (int s = W >> 1; s > 0; s >>= 1) {
    if (tid < s) red[tid] += red[tid + s];
    __syncthreads();
  }
  const float var = red[0] / (float)W - mean * mean;
  const float r = rsqrtf(var + 1e-5f);
  const float o = fmaxf((v - mean) * r * g[tid] + b[tid], 0.0f);
  if (WRITE_F32) Of[(size_t)n * W + tid] = o;
  if (WRITE_F16) Oh[(size_t)n * W + tid] = (_Float16)o;
}

// gate[level] = (mean_n clip(sigmoid(y2ln[n].pow_w + pow_b)) >= 0.5)
__global__ __launch_bounds__(256) void policy_gate(
    const float* __restrict__ Y, const float* __restrict__ pw,
    const float* __restrict__ pb, float* __restrict__ gates, int level) {
  __shared__ float red[256];
  const int tid = threadIdx.x;
  float local = 0.0f;
  for (int n = tid; n < NTOK; n += 256) {
    const float* row = Y + (size_t)n * 128;
    float z = pb[0];
    for (int k = 0; k < 128; ++k) z += row[k] * pw[k];
    float pr = 1.0f / (1.0f + expf(-z));
    pr = fminf(fmaxf(pr, 1e-7f), 1.0f - 1e-7f);
    local += pr;
  }
  red[tid] = local;
  __syncthreads();
  for (int s = 128; s > 0; s >>= 1) {
    if (tid < s) red[tid] += red[tid + s];
    __syncthreads();
  }
  if (tid == 0) gates[level] = (red[0] * (1.0f / NTOK) >= 0.5f) ? 1.0f : 0.0f;
}

__global__ void pooled_combine(const float* __restrict__ S0,
                               const float* __restrict__ S1,
                               const float* __restrict__ S2,
                               const float* __restrict__ S3,
                               const float* __restrict__ gates,
                               _Float16* __restrict__ pooled) {
  const float g1 = gates[0];
  const float g2 = g1 * gates[1];
  const float g3 = g2 * gates[2];
  const float inv = 1.0f / (1.0f + 2.0f * g1 + 4.0f * g2 + 8.0f * g3);
  const size_t i = (size_t)blockIdx.x * blockDim.x + threadIdx.x;  // 4096*512
  pooled[i] = (_Float16)((S0[i] + g1 * S1[i] + g2 * S2[i] + g3 * S3[i]) * inv);
}

// ---------------------------------------------------------------------------
extern "C" void kernel_launch(void* const* d_in, const int* in_sizes, int n_in,
                              void* d_out, int out_size, void* d_ws,
                              size_t ws_size, hipStream_t stream) {
  const int*   tok   = (const int*)d_in[0];
  const float* emb   = (const float*)d_in[1];
  const float* projw = (const float*)d_in[2];
  const float* projb = (const float*)d_in[3];
  const float* cfw   = (const float*)d_in[4];
  const float* cfb   = (const float*)d_in[5];
  const float* ing   = (const float*)d_in[6];
  const float* inb   = (const float*)d_in[7];
  const float* pde   = (const float*)d_in[8];
  const float* f1w   = (const float*)d_in[9];
  const float* f1b   = (const float*)d_in[10];
  const float* n1g   = (const float*)d_in[11];
  const float* n1b   = (const float*)d_in[12];
  const float* f2w   = (const float*)d_in[13];
  const float* f2b   = (const float*)d_in[14];
  const float* n2g   = (const float*)d_in[15];
  const float* n2b   = (const float*)d_in[16];
  const float* poww  = (const float*)d_in[17];
  const float* powb  = (const float*)d_in[18];
  const float* sib   = (const float*)d_in[19];
  const float* dep   = (const float*)d_in[20];
  const float* outw  = (const float*)d_in[21];
  const float* outb  = (const float*)d_in[22];
  float* out = (float*)d_out;

  // workspace carve-out (~105 MB total)
  char* p = (char*)d_ws;
  auto carve = [&](size_t bytes) -> char* {
    char* r = p;
    p += (bytes + 255) & ~(size_t)255;
    return r;
  };
  _Float16* projw_h = (_Float16*)carve((size_t)HDIM * HDIM * 2);
  _Float16* wc_h    = (_Float16*)carve((size_t)HDIM * HDIM * 2);
  _Float16* f1w_h   = (_Float16*)carve((size_t)256 * 640 * 2);
  _Float16* f2w_h   = (_Float16*)carve((size_t)128 * 256 * 2);
  _Float16* outw_h  = (_Float16*)carve((size_t)VDIM * HDIM * 2);
  float*    cvec    = (float*)carve((size_t)HDIM * 4);
  _Float16* A0_h    = (_Float16*)carve((size_t)NTOK * HDIM * 2);
  float*    Sf[4];
  for (int l = 0; l < 4; ++l) Sf[l] = (float*)carve((size_t)NTOK * HDIM * 4);
  _Float16* Sh[3];
  for (int l = 0; l < 3; ++l) Sh[l] = (_Float16*)carve((size_t)NTOK * HDIM * 2);
  _Float16* X_h    = (_Float16*)carve((size_t)NTOK * 640 * 2);
  float*    y1f    = (float*)carve((size_t)NTOK * 256 * 4);
  _Float16* y1h    = (_Float16*)carve((size_t)NTOK * 256 * 2);
  float*    y2f    = (float*)carve((size_t)NTOK * 128 * 4);
  float*    y2lf   = (float*)carve((size_t)NTOK * 128 * 4);
  _Float16* pool_h = (_Float16*)carve((size_t)NTOK * HDIM * 2);
  float*    gates  = (float*)carve(256);

  auto gemm_ff = [&](const _Float16* A, const _Float16* W, const float* bias,
                     float bscale, float* Cf, _Float16* Ch, int M, int N, int K) {
    dim3 grid(N / BN, M / BM);
    gemm_f16_wmma<true, true><<<grid, dim3(128), 0, stream>>>(
        A, W, bias, bscale, Cf, Ch, M, N, K);
  };
  auto gemm_f = [&](const _Float16* A, const _Float16* W, const float* bias,
                    float bscale, float* Cf, int M, int N, int K) {
    dim3 grid(N / BN, M / BM);
    gemm_f16_wmma<true, false><<<grid, dim3(128), 0, stream>>>(
        A, W, bias, bscale, Cf, nullptr, M, N, K);
  };

  // -- weight prep ----------------------------------------------------------
  f32_to_f16<<<512, 256, 0, stream>>>(projw, projw_h, (size_t)HDIM * HDIM);
  f32_to_f16<<<640, 256, 0, stream>>>(f1w, f1w_h, (size_t)256 * 640);
  f32_to_f16<<<128, 256, 0, stream>>>(f2w, f2w_h, (size_t)128 * 256);
  f32_to_f16<<<4096, 256, 0, stream>>>(outw, outw_h, (size_t)VDIM * HDIM);
  make_wc<<<HDIM * HDIM / 256, 256, 0, stream>>>(cfw, wc_h);
  make_cvec<<<2, 256, 0, stream>>>(cfb, sib, cvec);

  // -- h0 = emb[tok] @ proj_w^T + proj_b ------------------------------------
  gather_emb<<<NTOK * HDIM / 256, 256, 0, stream>>>(tok, emb, A0_h);
  gemm_ff(A0_h, projw_h, projb, 1.0f, Sf[0], Sh[0], NTOK, HDIM, HDIM);

  // -- 3 levels: policy gate + linear recurrence ----------------------------
  for (int l = 0; l < 3; ++l) {
    const float invM = 1.0f / (float)(1 << l);
    ln_policy_input<<<NTOK, 256, 0, stream>>>(Sf[l], invM, dep + l * HDIM, ing,
                                              inb, pde + l * 128, X_h);
    gemm_f(X_h, f1w_h, f1b, 1.0f, y1f, NTOK, 256, 640);
    ln_relu<false, true><<<NTOK, 256, 0, stream>>>(y1f, n1g, n1b, nullptr, y1h, 256);
    gemm_f(y1h, f2w_h, f2b, 1.0f, y2f, NTOK, 128, 256);
    ln_relu<true, false><<<NTOK, 128, 0, stream>>>(y2f, n2g, n2b, y2lf, nullptr, 128);
    policy_gate<<<1, 256, 0, stream>>>(y2lf, poww, powb, gates, l);
    if (l < 2)
      gemm_ff(Sh[l], wc_h, cvec, (float)(1 << l), Sf[l + 1], Sh[l + 1], NTOK,
              HDIM, HDIM);
    else
      gemm_f(Sh[l], wc_h, cvec, (float)(1 << l), Sf[l + 1], NTOK, HDIM, HDIM);
  }

  // -- gated mean pool + output projection ----------------------------------
  pooled_combine<<<NTOK * HDIM / 256, 256, 0, stream>>>(Sf[0], Sf[1], Sf[2],
                                                        Sf[3], gates, pool_h);
  gemm_f(pool_h, outw_h, outb, 1.0f, out, NTOK, VDIM, HDIM);
}
